// NystromLayer_29549374997067
// MI455X (gfx1250) — matrix operand order repaired and checked by reading, not compiled, not run on hardware
//
#include <hip/hip_runtime.h>
#include <hip/hip_bf16.h>
#include <cstddef>

// ---------------- problem constants ----------------
#define BATCH   16
#define SEQ     6000
#define DIM     512
#define HEADS   6
#define DH      64
#define INNER   384           // HEADS*DH
#define QKVCOLS 1152          // 3*INNER
#define NLAND   20
#define LSEG    300           // SEQ / NLAND
#define KER     33
#define KPAD    16
#define SCALE   0.125f        // DH^-0.5
#define ROWS    (BATCH*SEQ)   // 96000
#define BH      (BATCH*HEADS) // 96

typedef __attribute__((ext_vector_type(16))) _Float16 v16h;
typedef __attribute__((ext_vector_type(8)))  _Float16 v8h;
typedef __attribute__((ext_vector_type(4)))  _Float16 v4h;
typedef __attribute__((ext_vector_type(8)))  float    v8f;
typedef __attribute__((ext_vector_type(4)))  float    v4f;

__device__ __forceinline__ v8f wmma16(v16h a, v16h b, v8f c) {
  return __builtin_amdgcn_wmma_f32_16x16x32_f16(
      /*neg_a=*/false, a, /*neg_b=*/false, b,
      /*c_mod=*/(short)0, c, /*reuse_a=*/false, /*reuse_b=*/false);
}

// 16 consecutive halfs (two aligned 16B chunks)
__device__ __forceinline__ v16h ld16(const _Float16* p) {
  v8h lo = *(const v8h*)p;
  v8h hi = *(const v8h*)(p + 8);
  return __builtin_shufflevector(lo, hi, 0, 1, 2, 3, 4, 5, 6, 7,
                                 8, 9, 10, 11, 12, 13, 14, 15);
}
// two aligned 16B chunks 32B apart (A-layout: K base..base+7, base+16..base+23)
__device__ __forceinline__ v16h ld16_split(const _Float16* p) {
  v8h lo = *(const v8h*)p;
  v8h hi = *(const v8h*)(p + 16);
  return __builtin_shufflevector(lo, hi, 0, 1, 2, 3, 4, 5, 6, 7,
                                 8, 9, 10, 11, 12, 13, 14, 15);
}

// ---------------- LayerNorm stats ----------------
__global__ __launch_bounds__(256) void lnstats_kernel(
    const float* __restrict__ x, float* __restrict__ mu, float* __restrict__ rstd) {
  __shared__ float rs[256], rs2[256];
  const int row = blockIdx.x, tid = threadIdx.x;
  const float* xr = x + (size_t)row * DIM;
  float s = 0.f, s2 = 0.f;
  for (int j = tid; j < DIM; j += 256) { float v = xr[j]; s += v; s2 += v * v; }
  rs[tid] = s; rs2[tid] = s2; __syncthreads();
  for (int st = 128; st > 0; st >>= 1) {
    if (tid < st) { rs[tid] += rs[tid + st]; rs2[tid] += rs2[tid + st]; }
    __syncthreads();
  }
  if (tid == 0) {
    float m = rs[0] * (1.0f / DIM);
    float var = rs2[0] * (1.0f / DIM) - m * m;
    mu[row] = m;
    rstd[row] = rsqrtf(var + 1e-5f);
  }
}

// ---------------- GEMM tiles ----------------
// block tile 128x64, 8 waves (4x2), wave tile 32x32 (2x2 WMMA 16x16x32)
#define BM 128
#define BN 64
#define BK 32
#define LPAD 8   // row stride 40 halfs = 80B: 16B-aligned rows, conflict-free frag reads

// ---------------- GEMM1: qkv = LN(x) @ w_qkv (f16 WMMA, f32 accum) ----------------
__global__ __launch_bounds__(256) void gemm_qkv_kernel(
    const float* __restrict__ x, const float* __restrict__ mu,
    const float* __restrict__ rstd, const float* __restrict__ nw,
    const float* __restrict__ nb, const float* __restrict__ wqkv,
    float* __restrict__ qkv) {
  __shared__ __align__(16) _Float16 As[BM][BK + LPAD];   // row-major [m][k]
  __shared__ __align__(16) _Float16 Bs[BN][BK + LPAD];   // TRANSPOSED  [n][k]
  const int tid  = threadIdx.x;
  const int lane = tid & 31;
  const int wave = tid >> 5;
  const int wr = wave >> 1, wc = wave & 1;      // 4x2 wave grid
  const int bm = blockIdx.y * BM;
  const int bn = blockIdx.x * BN;
  const int K = DIM, NCOL = QKVCOLS;

  v8f c00 = {}, c01 = {}, c10 = {}, c11 = {};
  for (int k0 = 0; k0 < K; k0 += BK) {
    // A tile: float4 global loads, LayerNorm on the fly, f32 -> f16
#pragma unroll
    for (int j = 0; j < 4; ++j) {                 // 128x32 = 1024 float4 chunks
      int idx = tid + j * 256;
      int r = idx >> 3, c4 = (idx & 7) * 4;
      int row = bm + r, col = k0 + c4;
      v4f xv = *(const v4f*)&x[(size_t)row * K + col];
      v4f wv = *(const v4f*)&nw[col];
      v4f bv = *(const v4f*)&nb[col];
      float m_ = mu[row], rs_ = rstd[row];
      v4h hv;
#pragma unroll
      for (int e = 0; e < 4; ++e) hv[e] = (_Float16)((xv[e] - m_) * rs_ * wv[e] + bv[e]);
      *(v4h*)&As[r][c4] = hv;
    }
    // B tile: float4 global loads, scattered transposed LDS stores
#pragma unroll
    for (int j = 0; j < 2; ++j) {                 // 32x64 = 512 float4 chunks
      int idx = tid + j * 256;
      int r = idx >> 4, c4 = (idx & 15) * 4;
      v4f wv = *(const v4f*)&wqkv[(size_t)(k0 + r) * NCOL + bn + c4];
#pragma unroll
      for (int e = 0; e < 4; ++e) Bs[c4 + e][r] = (_Float16)wv[e];
    }
    // prefetch next A k-tile (global_prefetch_b8) while we compute
    if (k0 + BK < K) {
      int pr = tid >> 1, pc = (tid & 1) * 16;
      __builtin_prefetch(&x[(size_t)(bm + pr) * K + k0 + BK + pc], 0, 3);
    }
    __syncthreads();

    const int mrow  = lane & 15;
    const int abase = (lane >> 4) * 8;   // A: upper half-wave holds K+8
    const int bbase = (lane >> 4) * 8;   // B: upper half-wave holds K+8
    const int mW = wr * 32, nW = wc * 32;
    v16h a0 = ld16_split(&As[mW + mrow][abase]);
    v16h a1 = ld16_split(&As[mW + 16 + mrow][abase]);
    v16h b0 = ld16_split(&Bs[nW + mrow][bbase]);
    v16h b1 = ld16_split(&Bs[nW + 16 + mrow][bbase]);
    c00 = wmma16(a0, b0, c00);
    c01 = wmma16(a0, b1, c01);
    c10 = wmma16(a1, b0, c10);
    c11 = wmma16(a1, b1, c11);
    __syncthreads();
  }

  // epilogue: D layout -> qkv row-major, scale q columns
  const int nloc  = lane & 15;
  const int rhalf = (lane >> 4) * 8;
#pragma unroll
  for (int t = 0; t < 4; ++t) {
    v8f acc = (t == 0) ? c00 : (t == 1) ? c01 : (t == 2) ? c10 : c11;
    int mbase = wr * 32 + ((t >> 1) * 16);
    int nbase = wc * 32 + ((t & 1) * 16);
    int col = bn + nbase + nloc;
    float sc = (col < INNER) ? SCALE : 1.0f;   // q columns pre-scaled
#pragma unroll
    for (int r = 0; r < 8; ++r) {
      int row = bm + mbase + rhalf + r;
      qkv[(size_t)row * NCOL + col] = acc[r] * sc;
    }
  }
}

// ---------------- landmark means (q already scaled) ----------------
__global__ __launch_bounds__(256) void landmarks_kernel(
    const float* __restrict__ qkv, float* __restrict__ qland, float* __restrict__ kland) {
  int idx = blockIdx.x * 256 + threadIdx.x;        // 2 * BH*NLAND*DH = 245760
  int half = idx / (BH * NLAND * DH);
  int r = idx % (BH * NLAND * DH);
  int d = r & 63;
  int m = (r >> 6) % NLAND;
  int bh = r / (DH * NLAND);
  int b = bh / HEADS, h = bh % HEADS;
  int off = half ? INNER : 0;                      // 0:q 384:k
  const float* base = qkv + ((size_t)(b * SEQ + m * LSEG)) * QKVCOLS + off + h * DH + d;
  float s = 0.f;
  for (int i = 0; i < LSEG; ++i) s += base[(size_t)i * QKVCOLS];
  (half ? kland : qland)[bh * (NLAND * DH) + m * DH + d] = s * (1.0f / LSEG);
}

// ---------------- attn2 = softmax(q_land @ k_land^T), global row/col-sum maxes ----------------
__global__ __launch_bounds__(256) void attn2_kernel(
    const float* __restrict__ qland, const float* __restrict__ kland,
    float* __restrict__ attn2, float* __restrict__ stats) {
  __shared__ float ql[NLAND][DH], kl[NLAND][DH], s[NLAND][NLAND], red[2 * NLAND];
  const int bh = blockIdx.x, tid = threadIdx.x;
  for (int j = tid; j < NLAND * DH; j += 256) {
    ql[j >> 6][j & 63] = qland[bh * (NLAND * DH) + j];
    kl[j >> 6][j & 63] = kland[bh * (NLAND * DH) + j];
  }
  __syncthreads();
  for (int e = tid; e < NLAND * NLAND; e += 256) {
    int i = e / NLAND, j = e % NLAND;
    float acc = 0.f;
#pragma unroll
    for (int d = 0; d < DH; ++d) acc += ql[i][d] * kl[j][d];
    s[i][j] = acc;
  }
  __syncthreads();
  if (tid < NLAND) {
    int i = tid;
    float mx = -1e30f;
    for (int j = 0; j < NLAND; ++j) mx = fmaxf(mx, s[i][j]);
    float sum = 0.f;
    for (int j = 0; j < NLAND; ++j) { float e_ = expf(s[i][j] - mx); s[i][j] = e_; sum += e_; }
    float inv = 1.0f / sum;
    for (int j = 0; j < NLAND; ++j) s[i][j] *= inv;
  }
  __syncthreads();
  for (int e = tid; e < NLAND * NLAND; e += 256)
    attn2[bh * (NLAND * NLAND) + e] = s[e / NLAND][e % NLAND];
  if (tid < NLAND) {
    float cs = 0.f, rs = 0.f;
    for (int i = 0; i < NLAND; ++i) { cs += s[i][tid]; rs += s[tid][i]; }
    red[tid] = rs; red[NLAND + tid] = cs;
  }
  __syncthreads();
  if (tid == 0) {
    float rmax = 0.f, cmax = 0.f;
    for (int i = 0; i < NLAND; ++i) { rmax = fmaxf(rmax, red[i]); cmax = fmaxf(cmax, red[NLAND + i]); }
    atomicMax((int*)&stats[0], __float_as_int(rmax));   // positive floats: int order == float order
    atomicMax((int*)&stats[1], __float_as_int(cmax));
  }
}

// ---------------- sim3 = q_land @ k^T  (materialized [BH,20,SEQ]) ----------------
__global__ __launch_bounds__(256) void sim3_kernel(
    const float* __restrict__ qkv, const float* __restrict__ qland, float* __restrict__ sim3) {
  const int bh = blockIdx.x, chunk = blockIdx.y, tid = threadIdx.x;
  const int b = bh / HEADS, h = bh % HEADS;
  __shared__ float ql[NLAND][DH];
  for (int j = tid; j < NLAND * DH; j += 256) ql[j >> 6][j & 63] = qland[bh * (NLAND * DH) + j];
  __syncthreads();
  int n = chunk * 256 + tid;
  if (n >= SEQ) return;
  const float* kr = qkv + ((size_t)(b * SEQ + n)) * QKVCOLS + INNER + h * DH;
  float kv[DH];
#pragma unroll
  for (int d = 0; d < DH; ++d) kv[d] = kr[d];
#pragma unroll
  for (int i = 0; i < NLAND; ++i) {
    float acc = 0.f;
#pragma unroll
    for (int d = 0; d < DH; ++d) acc += ql[i][d] * kv[d];
    sim3[((size_t)bh * NLAND + i) * SEQ + n] = acc;
  }
}

// ---------------- attn3v = softmax(sim3) @ v  (one block per (b,h)) ----------------
__global__ __launch_bounds__(256) void a3v_kernel(
    const float* __restrict__ qkv, const float* __restrict__ sim3, float* __restrict__ attn3v) {
  constexpr int CH = 60, NCHUNK = SEQ / CH;      // 100 chunks
  __shared__ float vs[CH][DH];
  __shared__ float wts[NLAND][CH];
  __shared__ float mx[NLAND];
  __shared__ float red[256];
  __shared__ float ss[NLAND];
  const int bh = blockIdx.x, tid = threadIdx.x;
  const int b = bh / HEADS, h = bh % HEADS;

  // row maxes over SEQ
  for (int i = 0; i < NLAND; ++i) {
    const float* row = sim3 + ((size_t)bh * NLAND + i) * SEQ;
    float m = -1e30f;
    for (int n = tid; n < SEQ; n += 256) m = fmaxf(m, row[n]);
    red[tid] = m; __syncthreads();
    for (int st = 128; st > 0; st >>= 1) {
      if (tid < st) red[tid] = fmaxf(red[tid], red[tid + st]);
      __syncthreads();
    }
    if (tid == 0) mx[i] = red[0];
    __syncthreads();
  }

  float acc[5] = {0, 0, 0, 0, 0};   // 1280 outputs / 256 threads
  float ssum = 0.f;
  for (int c = 0; c < NCHUNK; ++c) {
    int n0 = c * CH;
    for (int j = tid; j < CH * DH; j += 256) {
      int r = j >> 6, d = j & 63;
      vs[r][d] = qkv[((size_t)(b * SEQ + n0 + r)) * QKVCOLS + 2 * INNER + h * DH + d];
    }
    for (int j = tid; j < NLAND * CH; j += 256) {
      int i = j / CH, n = j % CH;
      wts[i][n] = expf(sim3[((size_t)bh * NLAND + i) * SEQ + n0 + n] - mx[i]);
    }
    __syncthreads();
    if (tid < NLAND) {
      float s_ = 0.f;
      for (int n = 0; n < CH; ++n) s_ += wts[tid][n];
      ssum += s_;
    }
#pragma unroll
    for (int k = 0; k < 5; ++k) {
      int e = tid + k * 256;
      int i = e >> 6, d = e & 63;
      float a = acc[k];
      for (int n = 0; n < CH; ++n) a += wts[i][n] * vs[n][d];
      acc[k] = a;
    }
    __syncthreads();
  }
  if (tid < NLAND) ss[tid] = ssum;
  __syncthreads();
#pragma unroll
  for (int k = 0; k < 5; ++k) {
    int e = tid + k * 256;
    int i = e >> 6;
    attn3v[bh * (NLAND * DH) + e] = acc[k] / ss[i];
  }
}

// ---------------- Newton-Schulz pinv + W2 = pinv @ attn3v ----------------
#define MM20(A_, B_, C_)                                                        \
  {                                                                             \
    for (int e = tid; e < NLAND * NLAND; e += 256) {                            \
      int i_ = e / NLAND, j_ = e % NLAND;                                       \
      float s_ = 0.f;                                                           \
      _Pragma("unroll") for (int kk = 0; kk < NLAND; ++kk)                      \
          s_ += A_[i_][kk] * B_[kk][j_];                                        \
      C_[i_][j_] = s_;                                                          \
    }                                                                           \
    __syncthreads();                                                            \
  }

__global__ __launch_bounds__(256) void pinv_w2_kernel(
    const float* __restrict__ attn2, const float* __restrict__ stats,
    const float* __restrict__ attn3v, float* __restrict__ w2) {
  __shared__ float a[NLAND][NLAND], z[NLAND][NLAND], az[NLAND][NLAND];
  __shared__ float t1[NLAND][NLAND], t2[NLAND][NLAND];
  const int bh = blockIdx.x, tid = threadIdx.x;
  for (int e = tid; e < NLAND * NLAND; e += 256) a[e / NLAND][e % NLAND] = attn2[bh * 400 + e];
  __syncthreads();
  const float dinv = 1.0f / (stats[0] * stats[1]);
  for (int e = tid; e < NLAND * NLAND; e += 256)
    z[e / NLAND][e % NLAND] = a[e % NLAND][e / NLAND] * dinv;    // z = a^T / denom
  __syncthreads();

  for (int it = 0; it < 6; ++it) {
    MM20(a, z, az);                                               // az = a@z
    for (int e = tid; e < 400; e += 256) {
      int i = e / NLAND, j = e % NLAND;
      t1[i][j] = (i == j ? 7.0f : 0.0f) - az[i][j];
    }
    __syncthreads();
    MM20(az, t1, t2);                                             // t2 = az@(7I-az)
    for (int e = tid; e < 400; e += 256) {
      int i = e / NLAND, j = e % NLAND;
      t2[i][j] = (i == j ? 15.0f : 0.0f) - t2[i][j];
    }
    __syncthreads();
    MM20(az, t2, t1);                                             // t1 = az@(15I-...)
    for (int e = tid; e < 400; e += 256) {
      int i = e / NLAND, j = e % NLAND;
      t1[i][j] = (i == j ? 13.0f : 0.0f) - t1[i][j];
    }
    __syncthreads();
    MM20(z, t1, t2);                                              // t2 = z@(13I-...)
    for (int e = tid; e < 400; e += 256) {
      int i = e / NLAND, j = e % NLAND;
      z[i][j] = 0.25f * t2[i][j];
    }
    __syncthreads();
  }

  // W2 = z @ attn3v   [20,20]@[20,64]
#pragma unroll
  for (int k = 0; k < 5; ++k) {
    int e = tid + k * 256;
    int i = e >> 6, d = e & 63;
    float s = 0.f;
#pragma unroll
    for (int j = 0; j < NLAND; ++j) s += z[i][j] * attn3v[bh * (NLAND * DH) + j * DH + d];
    w2[bh * (NLAND * DH) + e] = s;
  }
}

// ---------------- out = softmax(q @ k_land^T) @ W2 ----------------
__global__ __launch_bounds__(256) void attn1_out_kernel(
    const float* __restrict__ qkv, const float* __restrict__ kland,
    const float* __restrict__ w2, float* __restrict__ out_inner) {
  __shared__ float kl[NLAND][DH], W[NLAND][DH];
  const int bh = blockIdx.x, chunk = blockIdx.y, tid = threadIdx.x;
  const int b = bh / HEADS, h = bh % HEADS;
  for (int j = tid; j < NLAND * DH; j += 256) {
    kl[j >> 6][j & 63] = kland[bh * (NLAND * DH) + j];
    W[j >> 6][j & 63]  = w2[bh * (NLAND * DH) + j];
  }
  __syncthreads();
  int n = chunk * 256 + tid;
  if (n >= SEQ) return;
  const float* qr = qkv + ((size_t)(b * SEQ + n)) * QKVCOLS + h * DH;  // pre-scaled q
  float lg[NLAND];
#pragma unroll
  for (int i = 0; i < NLAND; ++i) lg[i] = 0.f;
  for (int d = 0; d < DH; ++d) {
    float qd = qr[d];
#pragma unroll
    for (int i = 0; i < NLAND; ++i) lg[i] += qd * kl[i][d];
  }
  float mxv = -1e30f;
#pragma unroll
  for (int i = 0; i < NLAND; ++i) mxv = fmaxf(mxv, lg[i]);
  float sum = 0.f;
#pragma unroll
  for (int i = 0; i < NLAND; ++i) { lg[i] = expf(lg[i] - mxv); sum += lg[i]; }
  float inv = 1.0f / sum;
#pragma unroll
  for (int i = 0; i < NLAND; ++i) lg[i] *= inv;
  float* orow = out_inner + ((size_t)(b * SEQ + n)) * INNER + h * DH;
  for (int d = 0; d < DH; ++d) {
    float s = 0.f;
#pragma unroll
    for (int i = 0; i < NLAND; ++i) s += lg[i] * W[i][d];
    orow[d] = s;
  }
}

// ---------------- depthwise conv residual over seq, accumulate into out_inner ----------------
__global__ __launch_bounds__(256) void conv_kernel(
    const float* __restrict__ qkv, const float* __restrict__ rk, float* __restrict__ out_inner) {
  constexpr int TN = 128;
  __shared__ float vs[TN + 2 * KPAD][DH];
  __shared__ float kw[KER];
  const int bh = blockIdx.x, chunk = blockIdx.y, tid = threadIdx.x;
  const int b = bh / HEADS, h = bh % HEADS;
  const int n0 = chunk * TN;
  if (tid < KER) kw[tid] = rk[h * KER + tid];
  for (int j = tid; j < (TN + 2 * KPAD) * DH; j += 256) {
    int r = j >> 6, d = j & 63;
    int gn = n0 - KPAD + r;
    vs[r][d] = (gn >= 0 && gn < SEQ)
                   ? qkv[((size_t)(b * SEQ + gn)) * QKVCOLS + 2 * INNER + h * DH + d]
                   : 0.f;
  }
  __syncthreads();
#pragma unroll
  for (int k = 0; k < (TN * DH) / 256; ++k) {
    int e = tid + k * 256;
    int ln = e >> 6, d = e & 63;
    int n = n0 + ln;
    if (n < SEQ) {
      float s = 0.f;
#pragma unroll
      for (int t = 0; t < KER; ++t) s += kw[t] * vs[ln + t][d];   // cross-correlation
      out_inner[((size_t)(b * SEQ + n)) * INNER + h * DH + d] += s;
    }
  }
}

// ---------------- GEMM2: d_out = x + out_inner @ w_out + b_out ----------------
__global__ __launch_bounds__(256) void gemm_out_kernel(
    const float* __restrict__ ai, const float* __restrict__ wout,
    const float* __restrict__ x, const float* __restrict__ bout,
    float* __restrict__ out) {
  __shared__ __align__(16) _Float16 As[BM][BK + LPAD];
  __shared__ __align__(16) _Float16 Bs[BN][BK + LPAD];   // transposed [n][k]
  const int tid  = threadIdx.x;
  const int lane = tid & 31;
  const int wave = tid >> 5;
  const int wr = wave >> 1, wc = wave & 1;
  const int bm = blockIdx.y * BM;
  const int bn = blockIdx.x * BN;
  const int K = INNER, NCOL = DIM;

  v8f c00 = {}, c01 = {}, c10 = {}, c11 = {};
  for (int k0 = 0; k0 < K; k0 += BK) {
    // A tile: float4 loads, f32 -> f16
#pragma unroll
    for (int j = 0; j < 4; ++j) {
      int idx = tid + j * 256;
      int r = idx >> 3, c4 = (idx & 7) * 4;
      v4f xv = *(const v4f*)&ai[(size_t)(bm + r) * K + k0 + c4];
      v4h hv;
#pragma unroll
      for (int e = 0; e < 4; ++e) hv[e] = (_Float16)xv[e];
      *(v4h*)&As[r][c4] = hv;
    }
    // B tile: float4 loads, transposed LDS stores
#pragma unroll
    for (int j = 0; j < 2; ++j) {
      int idx = tid + j * 256;
      int r = idx >> 4, c4 = (idx & 15) * 4;
      v4f wv = *(const v4f*)&wout[(size_t)(k0 + r) * NCOL + bn + c4];
#pragma unroll
      for (int e = 0; e < 4; ++e) Bs[c4 + e][r] = (_Float16)wv[e];
    }
    if (k0 + BK < K) {
      int pr = tid >> 1, pc = (tid & 1) * 16;
      __builtin_prefetch(&ai[(size_t)(bm + pr) * K + k0 + BK + pc], 0, 3);
    }
    __syncthreads();

    const int mrow  = lane & 15;
    const int abase = (lane >> 4) * 8;
    const int bbase = (lane >> 4) * 8;
    const int mW = wr * 32, nW = wc * 32;
    v16h a0 = ld16_split(&As[mW + mrow][abase]);
    v16h a1 = ld16_split(&As[mW + 16 + mrow][abase]);
    v16h b0 = ld16_split(&Bs[nW + mrow][bbase]);
    v16h b1 = ld16_split(&Bs[nW + 16 + mrow][bbase]);
    c00 = wmma16(a0, b0, c00);
    c01 = wmma16(a0, b1, c01);
    c10 = wmma16(a1, b0, c10);
    c11 = wmma16(a1, b1, c11);
    __syncthreads();
  }

  const int nloc  = lane & 15;
  const int rhalf = (lane >> 4) * 8;
#pragma unroll
  for (int t = 0; t < 4; ++t) {
    v8f acc = (t == 0) ? c00 : (t == 1) ? c01 : (t == 2) ? c10 : c11;
    int mbase = wr * 32 + ((t >> 1) * 16);
    int nbase = wc * 32 + ((t & 1) * 16);
    int col = bn + nbase + nloc;
    float bv = bout[col];
#pragma unroll
    for (int r = 0; r < 8; ++r) {
      int row = bm + mbase + rhalf + r;
      size_t oi = (size_t)row * NCOL + col;
      out[oi] = x[oi] + acc[r] + bv;                  // residual + bias
    }
  }
}

// ---------------- host ----------------
extern "C" void kernel_launch(void* const* d_in, const int* in_sizes, int n_in,
                              void* d_out, int out_size, void* d_ws, size_t ws_size,
                              hipStream_t stream) {
  const float* x      = (const float*)d_in[0];
  const float* norm_w = (const float*)d_in[1];
  const float* norm_b = (const float*)d_in[2];
  const float* w_qkv  = (const float*)d_in[3];
  const float* w_out  = (const float*)d_in[4];
  const float* b_out  = (const float*)d_in[5];
  const float* res_k  = (const float*)d_in[6];
  float* out = (float*)d_out;

  float* ws = (float*)d_ws;
  size_t o = 0;
  auto alloc = [&](size_t n) { float* p = ws + o; o += (n + 63) & ~(size_t)63; return p; };
  float* mu        = alloc(ROWS);
  float* rstd      = alloc(ROWS);
  float* qkv       = alloc((size_t)ROWS * QKVCOLS);       // 442 MB
  float* qland     = alloc((size_t)BH * NLAND * DH);
  float* kland     = alloc((size_t)BH * NLAND * DH);
  float* attn2     = alloc((size_t)BH * NLAND * NLAND);
  float* stats     = alloc(64);                            // [0]=rowmax [1]=colmax
  float* sim3      = alloc((size_t)BH * NLAND * SEQ);      // 46 MB
  float* attn3v    = alloc((size_t)BH * NLAND * DH);
  float* w2        = alloc((size_t)BH * NLAND * DH);
  float* out_inner = alloc((size_t)ROWS * INNER);          // 147 MB

  hipMemsetAsync(stats, 0, 2 * sizeof(float), stream);

  lnstats_kernel<<<ROWS, 256, 0, stream>>>(x, mu, rstd);

  gemm_qkv_kernel<<<dim3(QKVCOLS / BN, ROWS / BM), 256, 0, stream>>>(
      x, mu, rstd, norm_w, norm_b, w_qkv, qkv);

  landmarks_kernel<<<(2 * BH * NLAND * DH) / 256, 256, 0, stream>>>(qkv, qland, kland);

  attn2_kernel<<<BH, 256, 0, stream>>>(qland, kland, attn2, stats);

  sim3_kernel<<<dim3(BH, (SEQ + 255) / 256), 256, 0, stream>>>(qkv, qland, sim3);

  a3v_kernel<<<BH, 256, 0, stream>>>(qkv, sim3, attn3v);

  pinv_w2_kernel<<<BH, 256, 0, stream>>>(attn2, stats, attn3v, w2);

  attn1_out_kernel<<<dim3(BH, (SEQ + 255) / 256), 256, 0, stream>>>(qkv, kland, w2, out_inner);

  conv_kernel<<<dim3(BH, (SEQ + 127) / 128), 256, 0, stream>>>(qkv, res_k, out_inner);

  gemm_out_kernel<<<dim3(DIM / BN, ROWS / BM), 256, 0, stream>>>(
      out_inner, w_out, x, b_out, out);
}
